// EquivariantDecoderBlock_43593918054753
// MI455X (gfx1250) — matrix operand order repaired and checked by reading, not compiled, not run on hardware
//
#include <hip/hip_runtime.h>
#include <math.h>

typedef __bf16 bf16;
typedef __attribute__((ext_vector_type(16))) __bf16 v16bf;
typedef __attribute__((ext_vector_type(8)))  float  v8f;

#define HD 128
#define EPB 32   // rows (edges/nodes) per block in fused kernels

union FragBF { v16bf v; unsigned int u[8]; };

// ---- WMMA fragment loaders (contiguous k = 16*half + s, same order for A and B) --------------
__device__ __forceinline__ v16bf ld_frag_a(const bf16* base, int ld, int k0) {
  const int lane = threadIdx.x & 31;
  const int half = lane >> 4;
  const int row  = lane & 15;
  const unsigned int* rp = (const unsigned int*)(base + row * ld);
  FragBF f;
#pragma unroll
  for (int p = 0; p < 8; ++p) {
    int kb = k0 + half * 16 + 2 * p;
    f.u[p] = rp[kb >> 1];
  }
  return f.v;
}

__device__ __forceinline__ v16bf ld_frag_b(const bf16* wt, int kld, int n0, int k0) {
  const int lane = threadIdx.x & 31;
  const int half = lane >> 4;
  const int col  = lane & 15;
  const unsigned int* rp = (const unsigned int*)(wt + (size_t)(n0 + col) * kld);
  FragBF f;
#pragma unroll
  for (int p = 0; p < 8; ++p) {
    int kb = k0 + half * 16 + 2 * p;
    f.u[p] = rp[kb >> 1];
  }
  return f.v;
}

__device__ __forceinline__ v8f wmma_bf16(v16bf a, v16bf b, v8f c) {
  return __builtin_amdgcn_wmma_f32_16x16x32_bf16(false, a, false, b, (short)0, c, false, false);
}

// two 16x16 output tiles sharing each B fragment (independent accumulators
// interleave to hide WMMA RAW hazard NOPs and halve weight-fragment loads)
__device__ __forceinline__ void tile_gemm128_x2(const bf16* A0, const bf16* A1, int lda,
                                                const bf16* Wt, int kld,
                                                int n0, int kw0, v8f& acc0, v8f& acc1) {
#pragma unroll
  for (int kk = 0; kk < 128; kk += 32) {
    v16bf b = ld_frag_b(Wt, kld, n0, kw0 + kk);
    acc0 = wmma_bf16(ld_frag_a(A0, lda, kk), b, acc0);
    acc1 = wmma_bf16(ld_frag_a(A1, lda, kk), b, acc1);
  }
}

__device__ __forceinline__ float silu_f(float x) {
  return x * __builtin_amdgcn_rcpf(1.0f + __expf(-x));
}
__device__ __forceinline__ float sigm_f(float x) {
  return __builtin_amdgcn_rcpf(1.0f + __expf(-x));
}

// ---- converters -------------------------------------------------------------
__global__ void k_f2bf(const float* __restrict__ src, bf16* __restrict__ dst, int n) {
  int i = blockIdx.x * blockDim.x + threadIdx.x;
  if (i < n) dst[i] = (bf16)src[i];
}

// src f32 [K][128] row-major -> dst bf16 [128][K]
__global__ void k_transpose_bf(const float* __restrict__ src, bf16* __restrict__ dst, int K) {
  int i = blockIdx.x * blockDim.x + threadIdx.x;
  if (i < K * HD) {
    int n = i / K, k = i % K;
    dst[i] = (bf16)src[k * HD + n];
  }
}

// ---- node-level vector-weight MLP: vw = MLP2(s; vm) (hoisted from per-edge) -
__global__ void __launch_bounds__(256)
k_node_vw(const bf16* __restrict__ s_bf,
          const bf16* __restrict__ t_vm1, const float* __restrict__ vm_b1,
          const bf16* __restrict__ t_vm2, const float* __restrict__ vm_b2,
          float* __restrict__ vw, int N) {
  __shared__ __align__(16) bf16 SA[EPB * HD];
  __shared__ __align__(16) bf16 H1[EPB * HD];
  const int nBase = blockIdx.x * EPB;
  const int tid = threadIdx.x;
  for (int idx = tid; idx < EPB * 64; idx += blockDim.x) {
    int row = idx >> 6, w = idx & 63;
    int node = nBase + row; if (node > N - 1) node = N - 1;
    ((unsigned int*)SA)[idx] = ((const unsigned int*)(s_bf + (size_t)node * HD))[w];
  }
  __syncthreads();
  const int wave = tid >> 5, lane = tid & 31;
  const int half = lane >> 4, c = lane & 15;
  const int n0 = wave * 16;
  v8f acc0, acc1;
  {
    float b = vm_b1[n0 + c];
#pragma unroll
    for (int r = 0; r < 8; ++r) { acc0[r] = b; acc1[r] = b; }
  }
  tile_gemm128_x2(SA, SA + 16 * HD, HD, t_vm1, HD, n0, 0, acc0, acc1);
#pragma unroll
  for (int r = 0; r < 8; ++r) {
    int M = r + half * 8;
    H1[M * HD + n0 + c]        = (bf16)silu_f(acc0[r]);
    H1[(M + 16) * HD + n0 + c] = (bf16)silu_f(acc1[r]);
  }
  __syncthreads();
  {
    float b = vm_b2[n0 + c];
#pragma unroll
    for (int r = 0; r < 8; ++r) { acc0[r] = b; acc1[r] = b; }
  }
  tile_gemm128_x2(H1, H1 + 16 * HD, HD, t_vm2, HD, n0, 0, acc0, acc1);
#pragma unroll
  for (int r = 0; r < 8; ++r) {
    int M = r + half * 8;
    int node0 = nBase + M, node1 = nBase + M + 16;
    if (node0 < N) vw[(size_t)node0 * HD + n0 + c] = acc0[r];
    if (node1 < N) vw[(size_t)node1 * HD + n0 + c] = acc1[r];
  }
}

// ---- fused edge kernel (32 edges / block, dual 16-row tiles per wave) -------
__global__ void __launch_bounds__(256)
k_edge(const bf16* __restrict__ s_bf,
       const float* __restrict__ v,
       const float* __restrict__ pos,
       const int* __restrict__ edges, int E,
       const float* __restrict__ edge_w1_f,   // f32, for row 256 (dist_sq rank-1)
       const bf16* __restrict__ t_ew1, const float* __restrict__ eb1,
       const bf16* __restrict__ t_ew2, const float* __restrict__ eb2,
       const bf16* __restrict__ t_pu1, const float* __restrict__ pu_b1,
       const float* __restrict__ pu_w2, const float* __restrict__ pu_b2,
       const float* __restrict__ vw, const float* __restrict__ vp_w,
       float* __restrict__ m_s, float* __restrict__ m_v,
       float* __restrict__ pos_acc, float* __restrict__ counts) {
  __shared__ __align__(16) bf16 SI[EPB * HD], SJ[EPB * HD], H1[EPB * HD], PHI[EPB * HD];
  __shared__ int   li[EPB], lj[EPB];
  __shared__ float ldsq[EPB], ldirx[EPB], ldiry[EPB], ldirz[EPB];
  __shared__ float lposw[EPB], lposv[EPB][3];
  const int eBase = blockIdx.x * EPB;
  const int tid = threadIdx.x;
  if (tid < EPB) {
    int e = eBase + tid; if (e > E - 1) e = E - 1;
    int i = edges[e], j = edges[E + e];
    li[tid] = i; lj[tid] = j;
    float rx = pos[i * 3 + 0] - pos[j * 3 + 0];
    float ry = pos[i * 3 + 1] - pos[j * 3 + 1];
    float rz = pos[i * 3 + 2] - pos[j * 3 + 2];
    float dsq = rx * rx + ry * ry + rz * rz;
    float dist = sqrtf(dsq + 1e-6f);
    float inv = 1.0f / (dist + 1e-8f);
    ldsq[tid] = dsq; ldirx[tid] = rx * inv; ldiry[tid] = ry * inv; ldirz[tid] = rz * inv;
    lposw[tid] = pu_b2[0];
    lposv[tid][0] = 0.f; lposv[tid][1] = 0.f; lposv[tid][2] = 0.f;
  }
  __syncthreads();
  for (int idx = tid; idx < EPB * 64; idx += blockDim.x) {
    int row = idx >> 6, w = idx & 63;
    ((unsigned int*)SI)[idx] = ((const unsigned int*)(s_bf + (size_t)li[row] * HD))[w];
    ((unsigned int*)SJ)[idx] = ((const unsigned int*)(s_bf + (size_t)lj[row] * HD))[w];
  }
  __syncthreads();
  const int wave = tid >> 5, lane = tid & 31;
  const int half = lane >> 4, c = lane & 15;
  const int n0 = wave * 16;
  v8f acc0, acc1;
  // edge MLP layer 1: s_i @ W1[0:128] + s_j @ W1[128:256] + dist_sq*W1[256] + b1
  {
    float b = eb1[n0 + c];
#pragma unroll
    for (int r = 0; r < 8; ++r) { acc0[r] = b; acc1[r] = b; }
  }
  tile_gemm128_x2(SI, SI + 16 * HD, HD, t_ew1, 256, n0, 0, acc0, acc1);
  tile_gemm128_x2(SJ, SJ + 16 * HD, HD, t_ew1, 256, n0, 128, acc0, acc1);
  {
    float wl = edge_w1_f[256 * HD + n0 + c];
#pragma unroll
    for (int r = 0; r < 8; ++r) {
      int M = r + half * 8;
      acc0[r] += ldsq[M] * wl;
      acc1[r] += ldsq[M + 16] * wl;
    }
  }
#pragma unroll
  for (int r = 0; r < 8; ++r) {
    int M = r + half * 8;
    H1[M * HD + n0 + c]        = (bf16)silu_f(acc0[r]);
    H1[(M + 16) * HD + n0 + c] = (bf16)silu_f(acc1[r]);
  }
  __syncthreads();
  // edge MLP layer 2 -> phi_e ; scatter into m_s
  {
    float b = eb2[n0 + c];
#pragma unroll
    for (int r = 0; r < 8; ++r) { acc0[r] = b; acc1[r] = b; }
  }
  tile_gemm128_x2(H1, H1 + 16 * HD, HD, t_ew2, HD, n0, 0, acc0, acc1);
#pragma unroll
  for (int r = 0; r < 8; ++r) {
    int M = r + half * 8;
    float p0 = acc0[r], p1 = acc1[r];
    PHI[M * HD + n0 + c]        = (bf16)p0;
    PHI[(M + 16) * HD + n0 + c] = (bf16)p1;
    if (eBase + M < E)      atomicAdd(&m_s[(size_t)li[M] * HD + n0 + c], p0);
    if (eBase + M + 16 < E) atomicAdd(&m_s[(size_t)li[M + 16] * HD + n0 + c], p1);
  }
  __syncthreads();
  // pos-weight MLP: silu(phi @ pu_w1 + b1) . pu_w2  (dot reduced via LDS atomics)
  {
    float b = pu_b1[n0 + c];
#pragma unroll
    for (int r = 0; r < 8; ++r) { acc0[r] = b; acc1[r] = b; }
  }
  tile_gemm128_x2(PHI, PHI + 16 * HD, HD, t_pu1, HD, n0, 0, acc0, acc1);
  {
    float w2 = pu_w2[n0 + c];
#pragma unroll
    for (int r = 0; r < 8; ++r) {
      int M = r + half * 8;
      atomicAdd(&lposw[M],      silu_f(acc0[r]) * w2);
      atomicAdd(&lposw[M + 16], silu_f(acc1[r]) * w2);
    }
  }
  // vector messages: v[j]*vw[j] -> m_v (global atomics) and vp_w-weighted pos term
  for (int idx = tid; idx < EPB * HD; idx += blockDim.x) {
    int e = idx >> 7, h = idx & 127;
    if (eBase + e < E) {
      int jn = lj[e], in_ = li[e];
      float wvv = vw[(size_t)jn * HD + h];
      const float* vp = v + ((size_t)jn * HD + h) * 3;
      float* mp = m_v + ((size_t)in_ * HD + h) * 3;
      float vpw = vp_w[h];
      float m0 = vp[0] * wvv, m1 = vp[1] * wvv, m2 = vp[2] * wvv;
      atomicAdd(mp + 0, m0); atomicAdd(mp + 1, m1); atomicAdd(mp + 2, m2);
      atomicAdd(&lposv[e][0], m0 * vpw);
      atomicAdd(&lposv[e][1], m1 * vpw);
      atomicAdd(&lposv[e][2], m2 * vpw);
    }
  }
  __syncthreads();
  if (tid < EPB && eBase + tid < E) {
    int in_ = li[tid];
    float px = lposw[tid] * ldirx[tid] + lposv[tid][0];
    float py = lposw[tid] * ldiry[tid] + lposv[tid][1];
    float pz = lposw[tid] * ldirz[tid] + lposv[tid][2];
    atomicAdd(&pos_acc[in_ * 3 + 0], px);
    atomicAdd(&pos_acc[in_ * 3 + 1], py);
    atomicAdd(&pos_acc[in_ * 3 + 2], pz);
    atomicAdd(&counts[in_], 1.0f);
  }
}

// ---- node update (32 nodes / block): s_new, gate, v_new ---------------------
__global__ void __launch_bounds__(256)
k_node_update(const float* __restrict__ s,
              const bf16* __restrict__ s_bf,
              const float* __restrict__ v,
              const float* __restrict__ m_s,
              const float* __restrict__ m_v,
              const bf16* __restrict__ t_su1, const float* __restrict__ su_b1,
              const bf16* __restrict__ t_su2, const float* __restrict__ su_b2,
              const bf16* __restrict__ t_vg, const float* __restrict__ vg_b,
              float* __restrict__ out_s, float* __restrict__ out_v, int N) {
  __shared__ __align__(16) bf16 SA[EPB * HD], MS[EPB * HD], H1[EPB * HD], SN[EPB * HD];
  __shared__ float GATE[EPB * HD];
  const int nBase = blockIdx.x * EPB;
  const int tid = threadIdx.x;
  for (int idx = tid; idx < EPB * 64; idx += blockDim.x) {
    int row = idx >> 6, w = idx & 63;
    int node = nBase + row; if (node > N - 1) node = N - 1;
    ((unsigned int*)SA)[idx] = ((const unsigned int*)(s_bf + (size_t)node * HD))[w];
  }
  for (int idx = tid; idx < EPB * HD; idx += blockDim.x) {
    int row = idx >> 7, h = idx & 127;
    int node = nBase + row; if (node > N - 1) node = N - 1;
    MS[idx] = (bf16)m_s[(size_t)node * HD + h];
  }
  __syncthreads();
  const int wave = tid >> 5, lane = tid & 31;
  const int half = lane >> 4, c = lane & 15;
  const int n0 = wave * 16;
  v8f acc0, acc1;
  {
    float b = su_b1[n0 + c];
#pragma unroll
    for (int r = 0; r < 8; ++r) { acc0[r] = b; acc1[r] = b; }
  }
  tile_gemm128_x2(SA, SA + 16 * HD, HD, t_su1, 256, n0, 0, acc0, acc1);
  tile_gemm128_x2(MS, MS + 16 * HD, HD, t_su1, 256, n0, 128, acc0, acc1);
#pragma unroll
  for (int r = 0; r < 8; ++r) {
    int M = r + half * 8;
    H1[M * HD + n0 + c]        = (bf16)silu_f(acc0[r]);
    H1[(M + 16) * HD + n0 + c] = (bf16)silu_f(acc1[r]);
  }
  __syncthreads();
  {
    float b = su_b2[n0 + c];
#pragma unroll
    for (int r = 0; r < 8; ++r) { acc0[r] = b; acc1[r] = b; }
  }
  tile_gemm128_x2(H1, H1 + 16 * HD, HD, t_su2, HD, n0, 0, acc0, acc1);
#pragma unroll
  for (int r = 0; r < 8; ++r) {
    int M = r + half * 8;
    int node0 = nBase + M, node1 = nBase + M + 16;
    float sn0 = acc0[r] + ((node0 < N) ? s[(size_t)node0 * HD + n0 + c] : 0.f);
    float sn1 = acc1[r] + ((node1 < N) ? s[(size_t)node1 * HD + n0 + c] : 0.f);
    SN[M * HD + n0 + c]        = (bf16)sn0;
    SN[(M + 16) * HD + n0 + c] = (bf16)sn1;
    if (node0 < N) out_s[(size_t)node0 * HD + n0 + c] = sn0;
    if (node1 < N) out_s[(size_t)node1 * HD + n0 + c] = sn1;
  }
  __syncthreads();
  {
    float b = vg_b[n0 + c];
#pragma unroll
    for (int r = 0; r < 8; ++r) { acc0[r] = b; acc1[r] = b; }
  }
  tile_gemm128_x2(SN, SN + 16 * HD, HD, t_vg, HD, n0, 0, acc0, acc1);
#pragma unroll
  for (int r = 0; r < 8; ++r) {
    int M = r + half * 8;
    GATE[M * HD + n0 + c]        = sigm_f(acc0[r]);
    GATE[(M + 16) * HD + n0 + c] = sigm_f(acc1[r]);
  }
  __syncthreads();
  for (int idx = tid; idx < EPB * HD; idx += blockDim.x) {
    int row = idx >> 7, h = idx & 127;
    int node = nBase + row;
    if (node < N) {
      float g = GATE[row * HD + h];
      size_t o = ((size_t)node * HD + h) * 3;
      out_v[o + 0] = v[o + 0] + m_v[o + 0] * g;
      out_v[o + 1] = v[o + 1] + m_v[o + 1] * g;
      out_v[o + 2] = v[o + 2] + m_v[o + 2] * g;
    }
  }
}

// ---- position finalize ------------------------------------------------------
__global__ void k_pos(const float* __restrict__ pos, const float* __restrict__ pos_acc,
                      const float* __restrict__ counts, float* __restrict__ out_pos, int N) {
  int i = blockIdx.x * blockDim.x + threadIdx.x;
  if (i < N) {
    float cnt = fmaxf(counts[i], 1.0f);
    float rc = __builtin_amdgcn_rcpf(cnt);
#pragma unroll
    for (int cix = 0; cix < 3; ++cix) {
      float d = pos_acc[i * 3 + cix] * rc;
      d = fminf(fmaxf(d, -1.0f), 1.0f);
      out_pos[i * 3 + cix] = pos[i * 3 + cix] + d * 1e-5f;
    }
  }
}

extern "C" void kernel_launch(void* const* d_in, const int* in_sizes, int n_in,
                              void* d_out, int out_size, void* d_ws, size_t ws_size,
                              hipStream_t stream) {
  (void)n_in; (void)out_size; (void)ws_size;
  const float* s       = (const float*)d_in[0];
  const float* v       = (const float*)d_in[1];
  const float* pos     = (const float*)d_in[2];
  const int*   edges   = (const int*)d_in[3];
  const float* edge_w1 = (const float*)d_in[4];
  const float* edge_b1 = (const float*)d_in[5];
  const float* edge_w2 = (const float*)d_in[6];
  const float* edge_b2 = (const float*)d_in[7];
  const float* vm_w1   = (const float*)d_in[8];
  const float* vm_b1   = (const float*)d_in[9];
  const float* vm_w2   = (const float*)d_in[10];
  const float* vm_b2   = (const float*)d_in[11];
  const float* pu_w1   = (const float*)d_in[12];
  const float* pu_b1   = (const float*)d_in[13];
  const float* pu_w2   = (const float*)d_in[14];
  const float* pu_b2   = (const float*)d_in[15];
  const float* su_w1   = (const float*)d_in[16];
  const float* su_b1   = (const float*)d_in[17];
  const float* su_w2   = (const float*)d_in[18];
  const float* su_b2   = (const float*)d_in[19];
  const float* vp_w    = (const float*)d_in[20];
  const float* vg_w    = (const float*)d_in[21];
  const float* vg_b    = (const float*)d_in[22];

  const int N = in_sizes[0] / HD;
  const int E = in_sizes[3] / 2;

  // workspace carve-out
  char* p = (char*)d_ws;
  auto alloc = [&](size_t bytes) -> char* {
    char* r = p;
    p += (bytes + 255) & ~(size_t)255;
    return r;
  };
  bf16*  s_bf    = (bf16*)alloc((size_t)N * HD * sizeof(bf16));
  float* vw      = (float*)alloc((size_t)N * HD * sizeof(float));
  float* m_s     = (float*)alloc((size_t)N * HD * sizeof(float));
  float* m_v     = (float*)alloc((size_t)N * HD * 3 * sizeof(float));
  float* pos_acc = (float*)alloc((size_t)N * 3 * sizeof(float));
  float* counts  = (float*)alloc((size_t)N * sizeof(float));
  bf16*  t_ew1   = (bf16*)alloc((size_t)HD * 256 * sizeof(bf16));
  bf16*  t_ew2   = (bf16*)alloc((size_t)HD * HD * sizeof(bf16));
  bf16*  t_vm1   = (bf16*)alloc((size_t)HD * HD * sizeof(bf16));
  bf16*  t_vm2   = (bf16*)alloc((size_t)HD * HD * sizeof(bf16));
  bf16*  t_pu1   = (bf16*)alloc((size_t)HD * HD * sizeof(bf16));
  bf16*  t_su1   = (bf16*)alloc((size_t)HD * 256 * sizeof(bf16));
  bf16*  t_su2   = (bf16*)alloc((size_t)HD * HD * sizeof(bf16));
  bf16*  t_vg    = (bf16*)alloc((size_t)HD * HD * sizeof(bf16));

  float* out     = (float*)d_out;
  float* out_s   = out;
  float* out_v   = out + (size_t)N * HD;
  float* out_pos = out + (size_t)N * HD + (size_t)N * HD * 3;

  hipMemsetAsync(m_s,     0, (size_t)N * HD * sizeof(float), stream);
  hipMemsetAsync(m_v,     0, (size_t)N * HD * 3 * sizeof(float), stream);
  hipMemsetAsync(pos_acc, 0, (size_t)N * 3 * sizeof(float), stream);
  hipMemsetAsync(counts,  0, (size_t)N * sizeof(float), stream);

  const int thr = 256;
  k_f2bf<<<(N * HD + thr - 1) / thr, thr, 0, stream>>>(s, s_bf, N * HD);
  k_transpose_bf<<<(256 * HD + thr - 1) / thr, thr, 0, stream>>>(edge_w1, t_ew1, 256);
  k_transpose_bf<<<(HD * HD + thr - 1) / thr, thr, 0, stream>>>(edge_w2, t_ew2, HD);
  k_transpose_bf<<<(HD * HD + thr - 1) / thr, thr, 0, stream>>>(vm_w1, t_vm1, HD);
  k_transpose_bf<<<(HD * HD + thr - 1) / thr, thr, 0, stream>>>(vm_w2, t_vm2, HD);
  k_transpose_bf<<<(HD * HD + thr - 1) / thr, thr, 0, stream>>>(pu_w1, t_pu1, HD);
  k_transpose_bf<<<(256 * HD + thr - 1) / thr, thr, 0, stream>>>(su_w1, t_su1, 256);
  k_transpose_bf<<<(HD * HD + thr - 1) / thr, thr, 0, stream>>>(su_w2, t_su2, HD);
  k_transpose_bf<<<(HD * HD + thr - 1) / thr, thr, 0, stream>>>(vg_w, t_vg, HD);

  k_node_vw<<<(N + EPB - 1) / EPB, 256, 0, stream>>>(s_bf, t_vm1, vm_b1, t_vm2, vm_b2, vw, N);

  k_edge<<<(E + EPB - 1) / EPB, 256, 0, stream>>>(
      s_bf, v, pos, edges, E,
      edge_w1, t_ew1, edge_b1, t_ew2, edge_b2,
      t_pu1, pu_b1, pu_w2, pu_b2,
      vw, vp_w, m_s, m_v, pos_acc, counts);

  k_node_update<<<(N + EPB - 1) / EPB, 256, 0, stream>>>(
      s, s_bf, v, m_s, m_v,
      t_su1, su_b1, t_su2, su_b2, t_vg, vg_b,
      out_s, out_v, N);

  k_pos<<<(N + thr - 1) / thr, thr, 0, stream>>>(pos, pos_acc, counts, out_pos, N);
}
